// EdgeFeatureGCN_48163763257453
// MI455X (gfx1250) — matrix operand order repaired and checked by reading, not compiled, not run on hardware
//
#include <hip/hip_runtime.h>

// ---------------- types ----------------
typedef __attribute__((ext_vector_type(8)))  float          v8f;
typedef __attribute__((ext_vector_type(16))) __bf16         v16bf;
typedef __attribute__((ext_vector_type(16))) unsigned short v16u;
typedef __attribute__((ext_vector_type(8)))  unsigned short v8u;

// ---------------- bf16 helpers ----------------
static __device__ __forceinline__ unsigned short f2bf_u(float f) {
  __bf16 b = (__bf16)f;                 // native f32->bf16 (RNE)
  unsigned short u; __builtin_memcpy(&u, &b, 2); return u;
}
static __device__ __forceinline__ float bf2f(unsigned short u) {
  union { unsigned u; float f; } c; c.u = ((unsigned)u) << 16; return c.f;
}

static __device__ __forceinline__ v8f wmma_bf16(v16bf a, v16bf b, v8f c) {
  // D = A(16x32 bf16) * B(32x16 bf16) + C(16x16 f32)
  return __builtin_amdgcn_wmma_f32_16x16x32_bf16(false, a, false, b, (short)0, c, false, false);
}

// A-fragment (16x32) from bf16 row-major data; rowbase = this lane's row + k0.
// ISA A layout: lane<16 holds K=k0..k0+7 & k0+16..k0+23; lane>=16 holds +8 of each.
static __device__ __forceinline__ v16bf a_frag_row(const unsigned short* rowbase, int lane) {
  const unsigned short* p = rowbase + ((lane >> 4) << 3);
  v8u lo = *(const v8u*)(p);
  v8u hi = *(const v8u*)(p + 16);
  v16u r;
  #pragma unroll
  for (int e = 0; e < 8; ++e) { r[e] = lo[e]; r[8 + e] = hi[e]; }
  return __builtin_bit_cast(v16bf, r);
}
// same but for a tiled LDS buffer where row = lane&15, stride ld (ld % 8 == 0)
static __device__ __forceinline__ v16bf a_frag_tile(const unsigned short* base, int ld,
                                                    int k0, int lane) {
  return a_frag_row(base + (lane & 15) * ld + k0, lane);
}

// ---------------- weight prep: f32 [K,N] row-major -> bf16 B-fragments ----------------
// frag element index = (tile_n*(Kpad/32) + ks)*512 + lane*16 + e
__global__ void prep_weight_kernel(const float* __restrict__ W, unsigned short* __restrict__ dst,
                                   int Kreal, int Kpad, int Np) {
  int tid = blockIdx.x * blockDim.x + threadIdx.x;
  int total = Kpad * Np;
  if (tid >= total) return;
  int blk = tid >> 9, r = tid & 511;
  int lane = r >> 4, e = r & 15;
  int nkk = Kpad >> 5;
  int tile_n = blk / nkk, ks = blk - tile_n * nkk;
  int n = tile_n * 16 + (lane & 15);
  int k = ks * 32 + ((lane >> 4) << 3) + (e & 7) + ((e >> 3) << 4);
  float v = (k < Kreal) ? W[k * Np + n] : 0.f;
  dst[tid] = f2bf_u(v);
}

// ---------------- data prep ----------------
__global__ void pad_x_kernel(const float* __restrict__ x, unsigned short* __restrict__ xb, int N) {
  int tid = blockIdx.x * blockDim.x + threadIdx.x;
  if (tid >= N * 32) return;
  int n = tid >> 5, k = tid & 31;
  xb[tid] = (k < 10) ? f2bf_u(x[n * 10 + k]) : (unsigned short)0;
}
__global__ void cvt_bf4_kernel(const float* __restrict__ src, unsigned short* __restrict__ dst,
                               long long total4) {
  long long t = (long long)blockIdx.x * blockDim.x + threadIdx.x;
  if (t >= total4) return;
  float4 f = *(const float4*)(src + t * 4);
  ushort4 u; u.x = f2bf_u(f.x); u.y = f2bf_u(f.y); u.z = f2bf_u(f.z); u.w = f2bf_u(f.w);
  *(ushort4*)(dst + t * 4) = u;
}

// ---------------- generic WMMA GEMM: act(A[MxK]_bf16 @ Bfrag + bias (+res_f32)) ----------------
__global__ __launch_bounds__(128) void gemm_bf16_kernel(
    const unsigned short* __restrict__ A, const unsigned short* __restrict__ Bfrag,
    const float* __restrict__ bias, const float* __restrict__ res,
    float* __restrict__ outF, unsigned short* __restrict__ outB,
    int M, int K, int Np, int relu) {
  int wv = threadIdx.x >> 5, lane = threadIdx.x & 31;
  int tiles_n = Np >> 4;
  int gid = blockIdx.x * (blockDim.x >> 5) + wv;
  int tile_m = gid / tiles_n, tile_n = gid - tile_m * tiles_n;
  if (tile_m * 16 >= M) return;
  int row = tile_m * 16 + (lane & 15);
  if (row >= M) row = M - 1;            // duplicate-load tail rows; stores are guarded
  const unsigned short* rowbase = A + (size_t)row * (size_t)K;
  int nk = K >> 5;
  const unsigned short* bp = Bfrag + (((size_t)tile_n * nk) << 9) + lane * 16;
  v8f c = {};
  for (int ks = 0; ks < nk; ++ks) {
    c = wmma_bf16(a_frag_row(rowbase + ks * 32, lane), *(const v16bf*)bp, c);
    bp += 512;
  }
  int n = tile_n * 16 + (lane & 15);
  float bv = bias ? bias[n] : 0.f;
  int mrow = tile_m * 16 + ((lane >> 4) << 3);
  #pragma unroll
  for (int r = 0; r < 8; ++r) {
    int m = mrow + r;
    if (m < M) {
      float v = c[r] + bv;
      if (res)  v += res[(size_t)m * Np + n];
      if (relu) v = fmaxf(v, 0.f);
      if (outF) outF[(size_t)m * Np + n] = v;
      if (outB) outB[(size_t)m * Np + n] = f2bf_u(v);
    }
  }
}

// ---------------- graph helper kernels ----------------
__global__ void fill_kernel(float* __restrict__ p, float v, int n) {
  int i = blockIdx.x * blockDim.x + threadIdx.x;
  if (i < n) p[i] = v;
}
__global__ void deg_kernel(const int* __restrict__ ei, float* __restrict__ deg, int E) {
  int e = blockIdx.x * blockDim.x + threadIdx.x;
  if (e < E) atomicAdd(&deg[ei[E + e]], 1.0f);
}
__global__ void rsqrt_kernel(float* __restrict__ d, int n) {
  int i = blockIdx.x * blockDim.x + threadIdx.x;
  if (i < n) d[i] = rsqrtf(d[i]);       // deg >= 1 (self loop)
}
__global__ void scatter_kernel(const int* __restrict__ ei, const float* __restrict__ dinv,
                               const float* __restrict__ hw, float* __restrict__ s, int E) {
  int t = blockIdx.x * blockDim.x + threadIdx.x;
  int e = t >> 5;
  if (e >= E) return;
  int c0 = (t & 31) << 2;
  int src = ei[e], dst = ei[E + e];
  float nrm = dinv[src] * dinv[dst];
  float4 hv = *(const float4*)(hw + (size_t)src * 128 + c0);
  float* o = s + (size_t)dst * 128 + c0;
  atomicAdd(o + 0, nrm * hv.x);
  atomicAdd(o + 1, nrm * hv.y);
  atomicAdd(o + 2, nrm * hv.z);
  atomicAdd(o + 3, nrm * hv.w);
}
__global__ void bn_kernel(float* __restrict__ s, const float* __restrict__ hw,
                          const float* __restrict__ dinv,
                          const float* __restrict__ cb, const float* __restrict__ g,
                          const float* __restrict__ b, const float* __restrict__ m,
                          const float* __restrict__ var, unsigned short* __restrict__ hb,
                          int total) {
  int idx = blockIdx.x * blockDim.x + threadIdx.x;
  if (idx >= total) return;
  int n = idx >> 7, c = idx & 127;
  float di = dinv[n];
  float base = s[idx] + di * di * hw[idx] + cb[c];
  float v = (base - m[c]) * rsqrtf(var[c] + 1e-5f) * g[c] + b[c];
  v = fmaxf(v, 0.f);
  s[idx] = v;
  if (hb) hb[idx] = f2bf_u(v);
}

// ---------------- fused per-edge MLP megakernel (persistent, LDS-staged weights) ----------------
#define EW 2                      // waves per block
#define TMPLD 136                 // 128-wide bf16 LDS tile stride (mult of 8)
#define RPLD  264                 // 256-wide bf16 LDS tile stride (mult of 8)
#define Z2LD  72
// edge-weight fragment pack offsets (bf16 elements) inside one contiguous buffer
#define OEE1 0
#define OEE2 16384
#define OEA1 32768
#define OCL1 81920
#define OCL2 114688
#define WTOT 122880               // total bf16 elements = 240 KB

__global__ __launch_bounds__(64) void edge_mlp_kernel(
    const int* __restrict__ ei, const unsigned short* __restrict__ hB,
    const unsigned short* __restrict__ eaB, const unsigned short* __restrict__ fEDGE,
    const float* __restrict__ ee1_b, const float* __restrict__ ee2_b,
    const float* __restrict__ ea1_b, const float* __restrict__ ea2_w,
    const float* __restrict__ ea2_b, const float* __restrict__ cl1_b,
    const float* __restrict__ cl2_b, const float* __restrict__ cl3_w,
    const float* __restrict__ cl3_b, float* __restrict__ out, int E) {
  __shared__ __align__(32) unsigned short wlds[WTOT];                         // 240 KB weights
  __shared__ __align__(32) unsigned short smem[EW * (16*TMPLD + 16*TMPLD + 16*RPLD)];
  __shared__ float sW[EW * 16];

  // ---- stage all edge-MLP weight fragments into LDS via async DMA ----
  {
    unsigned ldsbase = (unsigned)(unsigned long long)(const void*)&wlds[0];
    const int chunks = (WTOT * 2) / 16;                  // 16 B per issue, exact multiple of 64
    for (int i = threadIdx.x; i < chunks; i += 64) {
      unsigned lds = ldsbase + i * 16;
      const unsigned short* g = fEDGE + i * 8;
      asm volatile("global_load_async_to_lds_b128 %0, %1, off"
                   :: "v"(lds), "v"(g) : "memory");
    }
    asm volatile("s_wait_asynccnt 0x0" ::: "memory");    // wave's own copies done
    __syncthreads();                                     // publish LDS to both waves
  }
  const unsigned short* fEE1 = wlds + OEE1;
  const unsigned short* fEE2 = wlds + OEE2;
  const unsigned short* fEA1 = wlds + OEA1;
  const unsigned short* fCL1 = wlds + OCL1;
  const unsigned short* fCL2 = wlds + OCL2;

  int wv = threadIdx.x >> 5, lane = threadIdx.x & 31;
  unsigned short* sTmp = smem + wv * (16*TMPLD + 16*TMPLD + 16*RPLD);
  unsigned short* sE   = sTmp + 16 * TMPLD;     // e-values; later reused as z2
  unsigned short* sRp  = sE + 16 * TMPLD;       // edge_repr (16x256)
  float* wvw = sW + wv * 16;
  int half = lane >> 4, nl = lane & 15;

  for (int tile = blockIdx.x * EW + wv; tile * 16 < E; tile += (int)gridDim.x * EW) {
    int eid = tile * 16 + nl;
    int srcR = ei[eid], dstR = ei[E + eid];
    const unsigned short* eaRow = eaB + (size_t)eid * 128;
    const unsigned short* hsRow = hB + (size_t)srcR * 128;  // hr row for this lane's tile row
    const unsigned short* hdRow = hB + (size_t)dstR * 128;  // hc row
    __builtin_prefetch(hsRow, 0, 1);               // global_prefetch_b8
    __builtin_prefetch(hdRow, 0, 1);
    __builtin_prefetch(eaRow, 0, 1);

    // ---- t = relu(edge_attr @ ee1 + b) -> sTmp
    for (int nt = 0; nt < 8; ++nt) {
      v8f c = {};
      const unsigned short* bp = fEE1 + (((size_t)nt * 4) << 9) + lane * 16;
      #pragma unroll
      for (int ks = 0; ks < 4; ++ks) { c = wmma_bf16(a_frag_row(eaRow + ks * 32, lane), *(const v16bf*)bp, c); bp += 512; }
      float bv = ee1_b[nt * 16 + nl];
      #pragma unroll
      for (int r = 0; r < 8; ++r) sTmp[(r + half * 8) * TMPLD + nt * 16 + nl] = f2bf_u(fmaxf(c[r] + bv, 0.f));
    }
    // ---- e = t @ ee2 + b -> sE
    for (int nt = 0; nt < 8; ++nt) {
      v8f c = {};
      const unsigned short* bp = fEE2 + (((size_t)nt * 4) << 9) + lane * 16;
      #pragma unroll
      for (int ks = 0; ks < 4; ++ks) { c = wmma_bf16(a_frag_tile(sTmp, TMPLD, ks * 32, lane), *(const v16bf*)bp, c); bp += 512; }
      float bv = ee2_b[nt * 16 + nl];
      #pragma unroll
      for (int r = 0; r < 8; ++r) sE[(r + half * 8) * TMPLD + nt * 16 + nl] = f2bf_u(c[r] + bv);
    }
    // ---- a = relu([hr,hc,edge_attr] @ ea1 + b) -> sTmp  (K = 384)
    for (int nt = 0; nt < 8; ++nt) {
      v8f c = {};
      const unsigned short* bp = fEA1 + (((size_t)nt * 12) << 9) + lane * 16;
      #pragma unroll
      for (int ks = 0; ks < 4; ++ks) { c = wmma_bf16(a_frag_row(hsRow + ks * 32, lane), *(const v16bf*)bp, c); bp += 512; }
      #pragma unroll
      for (int ks = 0; ks < 4; ++ks) { c = wmma_bf16(a_frag_row(hdRow + ks * 32, lane), *(const v16bf*)bp, c); bp += 512; }
      #pragma unroll
      for (int ks = 0; ks < 4; ++ks) { c = wmma_bf16(a_frag_row(eaRow + ks * 32, lane), *(const v16bf*)bp, c); bp += 512; }
      float bv = ea1_b[nt * 16 + nl];
      #pragma unroll
      for (int r = 0; r < 8; ++r) sTmp[(r + half * 8) * TMPLD + nt * 16 + nl] = f2bf_u(fmaxf(c[r] + bv, 0.f));
    }
    // ---- gate w = sigmoid(a @ ea2 + b), one lane per edge row (4-wide)
    if (lane < 16) {
      float acc = ea2_b[0];
      for (int k = 0; k < 128; k += 4) {
        ushort4 u = *(const ushort4*)(sTmp + lane * TMPLD + k);
        float4 wv4 = *(const float4*)(ea2_w + k);
        acc += bf2f(u.x) * wv4.x + bf2f(u.y) * wv4.y + bf2f(u.z) * wv4.z + bf2f(u.w) * wv4.w;
      }
      wvw[lane] = 1.f / (1.f + __expf(-acc));
    }
    // ---- edge_repr = [hr + w*e, hc + w*e] -> sRp (same wave; LDS ops in-order)
    {
      int m = nl;
      float wm = wvw[m];
      const unsigned short* hx = half ? hdRow : hsRow;
      for (int k = 0; k < 128; k += 4) {
        ushort4 hu = *(const ushort4*)(hx + k);
        ushort4 eu = *(const ushort4*)(sE + m * TMPLD + k);
        ushort4 o;
        o.x = f2bf_u(bf2f(hu.x) + wm * bf2f(eu.x));
        o.y = f2bf_u(bf2f(hu.y) + wm * bf2f(eu.y));
        o.z = f2bf_u(bf2f(hu.z) + wm * bf2f(eu.z));
        o.w = f2bf_u(bf2f(hu.w) + wm * bf2f(eu.w));
        *(ushort4*)(sRp + m * RPLD + half * 128 + k) = o;
      }
    }
    // ---- z1 = relu(edge_repr @ cl1 + b) -> sTmp  (K = 256)
    for (int nt = 0; nt < 8; ++nt) {
      v8f c = {};
      const unsigned short* bp = fCL1 + (((size_t)nt * 8) << 9) + lane * 16;
      #pragma unroll
      for (int ks = 0; ks < 8; ++ks) { c = wmma_bf16(a_frag_tile(sRp, RPLD, ks * 32, lane), *(const v16bf*)bp, c); bp += 512; }
      float bv = cl1_b[nt * 16 + nl];
      #pragma unroll
      for (int r = 0; r < 8; ++r) sTmp[(r + half * 8) * TMPLD + nt * 16 + nl] = f2bf_u(fmaxf(c[r] + bv, 0.f));
    }
    // ---- z2 = relu(z1 @ cl2 + b) -> sE (reused, stride Z2LD)  N = 64
    for (int nt = 0; nt < 4; ++nt) {
      v8f c = {};
      const unsigned short* bp = fCL2 + (((size_t)nt * 4) << 9) + lane * 16;
      #pragma unroll
      for (int ks = 0; ks < 4; ++ks) { c = wmma_bf16(a_frag_tile(sTmp, TMPLD, ks * 32, lane), *(const v16bf*)bp, c); bp += 512; }
      float bv = cl2_b[nt * 16 + nl];
      #pragma unroll
      for (int r = 0; r < 8; ++r) sE[(r + half * 8) * Z2LD + nt * 16 + nl] = f2bf_u(fmaxf(c[r] + bv, 0.f));
    }
    // ---- out = z2 @ cl3 + b   (64 -> 2)
    if (lane < 16) {
      float a0 = cl3_b[0], a1 = cl3_b[1];
      for (int k = 0; k < 64; k += 2) {
        ushort2 u = *(const ushort2*)(sE + lane * Z2LD + k);
        float4 w4 = *(const float4*)(cl3_w + k * 2);
        float z0 = bf2f(u.x), z1 = bf2f(u.y);
        a0 += z0 * w4.x + z1 * w4.z;
        a1 += z0 * w4.y + z1 * w4.w;
      }
      out[(size_t)eid * 2 + 0] = a0;
      out[(size_t)eid * 2 + 1] = a1;
    }
  }
}

// ---------------- host launcher ----------------
extern "C" void kernel_launch(void* const* d_in, const int* in_sizes, int n_in,
                              void* d_out, int out_size, void* d_ws, size_t ws_size,
                              hipStream_t stream) {
  (void)in_sizes; (void)n_in; (void)out_size; (void)ws_size;
  const int N = 50000, E = 400000, H = 128, L = 4;

  const float* x      = (const float*)d_in[0];
  const int*   ei     = (const int*)d_in[1];
  const float* ea     = (const float*)d_in[2];
  const float* ne1_w  = (const float*)d_in[3];
  const float* ne1_b  = (const float*)d_in[4];
  const float* ne2_w  = (const float*)d_in[5];
  const float* ne2_b  = (const float*)d_in[6];
  const float* ee1_w  = (const float*)d_in[7];
  const float* ee1_b  = (const float*)d_in[8];
  const float* ee2_w  = (const float*)d_in[9];
  const float* ee2_b  = (const float*)d_in[10];
  const float* ea1_w  = (const float*)d_in[11];
  const float* ea1_b  = (const float*)d_in[12];
  const float* ea2_w  = (const float*)d_in[13];
  const float* ea2_b  = (const float*)d_in[14];
  const float* conv_w = (const float*)d_in[15];
  const float* conv_b = (const float*)d_in[16];
  const float* bn_g   = (const float*)d_in[17];
  const float* bn_b   = (const float*)d_in[18];
  const float* bn_m   = (const float*)d_in[19];
  const float* bn_v   = (const float*)d_in[20];
  const float* rp_w   = (const float*)d_in[21];
  const float* rp_b   = (const float*)d_in[22];
  const float* cl1_w  = (const float*)d_in[23];
  const float* cl1_b  = (const float*)d_in[24];
  const float* cl2_w  = (const float*)d_in[25];
  const float* cl2_b  = (const float*)d_in[26];
  const float* cl3_w  = (const float*)d_in[27];
  const float* cl3_b  = (const float*)d_in[28];
  float* out = (float*)d_out;

  // workspace carve-up
  char* ws = (char*)d_ws;
  size_t off = 0;
  auto walloc = [&](size_t bytes) {
    void* p = ws + off;
    off = (off + bytes + 255) & ~(size_t)255;
    return p;
  };
  size_t matF = (size_t)N * H * sizeof(float);
  size_t matB = (size_t)N * H * sizeof(unsigned short);
  float* hwF = (float*)walloc(matF);
  float* sF  = (float*)walloc(matF);
  unsigned short* Hbf[2];
  Hbf[0] = (unsigned short*)walloc(matB);
  Hbf[1] = (unsigned short*)walloc(matB);
  unsigned short* Xbf  = (unsigned short*)walloc((size_t)N * 32 * 2);
  unsigned short* eaBf = (unsigned short*)walloc((size_t)E * H * 2);
  float* dinv = (float*)walloc((size_t)N * sizeof(float));
  unsigned short* fNE1  = (unsigned short*)walloc((size_t)32 * 128 * 2);
  unsigned short* fNE2  = (unsigned short*)walloc((size_t)128 * 128 * 2);
  unsigned short* fCONV = (unsigned short*)walloc((size_t)L * 128 * 128 * 2);
  unsigned short* fRP   = (unsigned short*)walloc((size_t)(L - 1) * 128 * 128 * 2);
  unsigned short* fEDGE = (unsigned short*)walloc((size_t)WTOT * 2);   // packed edge weights

  auto prep = [&](const float* W, unsigned short* dst, int Kreal, int Kpad, int Np) {
    int total = Kpad * Np;
    prep_weight_kernel<<<(total + 255) / 256, 256, 0, stream>>>(W, dst, Kreal, Kpad, Np);
  };
  prep(ne1_w, fNE1, 10, 32, 128);
  prep(ne2_w, fNE2, 128, 128, 128);
  for (int l = 0; l < L; ++l)     prep(conv_w + (size_t)l * H * H, fCONV + (size_t)l * H * H, 128, 128, 128);
  for (int l = 0; l < L - 1; ++l) prep(rp_w + (size_t)l * H * H, fRP + (size_t)l * H * H, 128, 128, 128);
  prep(ee1_w, fEDGE + OEE1, 128, 128, 128);
  prep(ee2_w, fEDGE + OEE2, 128, 128, 128);
  prep(ea1_w, fEDGE + OEA1, 384, 384, 128);
  prep(cl1_w, fEDGE + OCL1, 256, 256, 128);
  prep(cl2_w, fEDGE + OCL2, 128, 128, 64);

  // data prep: x -> padded bf16, edge_attr -> bf16
  pad_x_kernel<<<(N * 32 + 255) / 256, 256, 0, stream>>>(x, Xbf, N);
  {
    long long t4 = (long long)E * H / 4;
    cvt_bf4_kernel<<<(int)((t4 + 255) / 256), 256, 0, stream>>>(ea, eaBf, t4);
  }

  auto gemm = [&](const unsigned short* A, const unsigned short* Bf, const float* bias,
                  const float* res, float* oF, unsigned short* oB,
                  int M, int K, int Np, int relu) {
    int tiles = ((M + 15) / 16) * (Np / 16);
    gemm_bf16_kernel<<<(tiles + 3) / 4, 128, 0, stream>>>(A, Bf, bias, res, oF, oB, M, K, Np, relu);
  };

  // GCN normalization: deg (with self loop) -> dinv
  fill_kernel<<<(N + 255) / 256, 256, 0, stream>>>(dinv, 1.0f, N);
  deg_kernel<<<(E + 255) / 256, 256, 0, stream>>>(ei, dinv, E);
  rsqrt_kernel<<<(N + 255) / 256, 256, 0, stream>>>(dinv, N);

  // node encoder: h = relu(x@ne1+b)@ne2+b  (h kept only as bf16)
  gemm(Xbf,    fNE1, ne1_b, nullptr, nullptr, Hbf[0], N, 32, 128, 1);
  gemm(Hbf[0], fNE2, ne2_b, nullptr, nullptr, Hbf[1], N, 128, 128, 0);
  int cur = 1;

  for (int l = 0; l < L; ++l) {
    gemm(Hbf[cur], fCONV + (size_t)l * H * H, nullptr, nullptr, hwF, nullptr, N, 128, 128, 0);
    hipMemsetAsync(sF, 0, matF, stream);
    {
      long long tt = (long long)E * 32;
      scatter_kernel<<<(int)((tt + 255) / 256), 256, 0, stream>>>(ei, dinv, hwF, sF, E);
    }
    bn_kernel<<<(N * H + 255) / 256, 256, 0, stream>>>(
        sF, hwF, dinv, conv_b + (size_t)l * H, bn_g + (size_t)l * H, bn_b + (size_t)l * H,
        bn_m + (size_t)l * H, bn_v + (size_t)l * H,
        (l == 0) ? Hbf[1 - cur] : nullptr, N * H);
    if (l == 0) {
      cur ^= 1;                               // h = bn result (bf16 copy)
    } else {
      gemm(Hbf[cur], fRP + (size_t)(l - 1) * H * H, rp_b + (size_t)(l - 1) * H, sF,
           nullptr, Hbf[1 - cur], N, 128, 128, 0);   // h = bn + h_res@rp + rp_b
      cur ^= 1;
    }
  }

  // fused edge pipeline -> out [E,2]  (persistent blocks, weights staged in LDS)
  edge_mlp_kernel<<<512, 64, 0, stream>>>(
      ei, Hbf[cur], eaBf, fEDGE,
      ee1_b, ee2_b, ea1_b, ea2_w, ea2_b, cl1_b, cl2_b, cl3_w, cl3_b, out, E);
}
